// GraphVAE_5162550690506
// MI455X (gfx1250) — compile-verified, matrix operand order checked by reading, not run on hardware
//
#include <hip/hip_runtime.h>

typedef __attribute__((ext_vector_type(16))) _Float16 v16h;
typedef __attribute__((ext_vector_type(8)))  float    v8f;

#define N_NODES 50000
#define N_EDGES 1600000
#define IN_CH 128
#define HID 128
#define LAT 32

// ---------------------------------------------------------------------------
// generic zero
__global__ void k_zero(float* p, int n) {
    int i = blockIdx.x * blockDim.x + threadIdx.x;
    if (i < n) p[i] = 0.0f;
}

// ---------------------------------------------------------------------------
// transpose + convert f32 W[K][N] -> f16 Wt[N][K]
__global__ void k_w_to_f16t(const float* __restrict__ w, _Float16* __restrict__ wt,
                            int K, int N) {
    int i = blockIdx.x * blockDim.x + threadIdx.x;   // over N*K outputs
    if (i >= K * N) return;
    int n = i / K;
    int k = i % K;
    wt[n * K + k] = (_Float16)w[k * N + n];
}

// ---------------------------------------------------------------------------
// edge aggregation: agg[dst] += feat[src], cnt[dst] += 1
// 32 threads per edge, 4 floats each (float4 gather + 4 global f32 atomics)
__global__ void k_aggregate(const float* __restrict__ feat,
                            const int* __restrict__ src,
                            const int* __restrict__ dst,
                            float* __restrict__ agg,
                            float* __restrict__ cnt,
                            int nEdges, int addCnt) {
    int gt = blockIdx.x * blockDim.x + threadIdx.x;
    if (gt >= nEdges * 32) return;
    int e    = gt >> 5;
    int part = gt & 31;
    int s = src[e];
    int d = dst[e];
    const float4 v = *(const float4*)(feat + (size_t)s * 128 + part * 4);
    float* o = agg + (size_t)d * 128 + part * 4;
    atomicAdd(o + 0, v.x);
    atomicAdd(o + 1, v.y);
    atomicAdd(o + 2, v.z);
    atomicAdd(o + 3, v.w);
    if (addCnt && part == 0) atomicAdd(cnt + d, 1.0f);
}

// ---------------------------------------------------------------------------
// fused SAGE layer: out = relu( (agg/max(cnt,1)) @ Wl + xin @ Wr + b )
// block = 256 threads = 8 waves; block handles a 16-row x 128-col strip;
// wave w owns columns [16w, 16w+16). Weights are f16, pre-transposed [col][k].
__global__ void __launch_bounds__(256)
k_sage_wmma(const float* __restrict__ agg,
            const float* __restrict__ cnt,
            const float* __restrict__ xin,
            const _Float16* __restrict__ WlT,
            const _Float16* __restrict__ WrT,
            const float* __restrict__ bias,
            float* __restrict__ out) {
    __shared__ _Float16 sMean[16][136];   // padded rows: 272B stride, 16B aligned
    __shared__ _Float16 sX[16][136];

    const int tid     = threadIdx.x;
    const int rowBase = blockIdx.x * 16;

    // cooperative stage: 2048 elems per matrix, 8 per thread
    {
        int base = tid * 8;
        int r = base >> 7;          // 0..15
        int k = base & 127;
        int grow = rowBase + r;
        float c  = cnt[grow];
        float inv = 1.0f / fmaxf(c, 1.0f);
        const float4* pa = (const float4*)(agg + (size_t)grow * 128 + k);
        float4 a0 = pa[0], a1 = pa[1];
        sMean[r][k + 0] = (_Float16)(a0.x * inv);
        sMean[r][k + 1] = (_Float16)(a0.y * inv);
        sMean[r][k + 2] = (_Float16)(a0.z * inv);
        sMean[r][k + 3] = (_Float16)(a0.w * inv);
        sMean[r][k + 4] = (_Float16)(a1.x * inv);
        sMean[r][k + 5] = (_Float16)(a1.y * inv);
        sMean[r][k + 6] = (_Float16)(a1.z * inv);
        sMean[r][k + 7] = (_Float16)(a1.w * inv);
        const float4* px = (const float4*)(xin + (size_t)grow * 128 + k);
        float4 x0 = px[0], x1 = px[1];
        sX[r][k + 0] = (_Float16)x0.x;
        sX[r][k + 1] = (_Float16)x0.y;
        sX[r][k + 2] = (_Float16)x0.z;
        sX[r][k + 3] = (_Float16)x0.w;
        sX[r][k + 4] = (_Float16)x1.x;
        sX[r][k + 5] = (_Float16)x1.y;
        sX[r][k + 6] = (_Float16)x1.z;
        sX[r][k + 7] = (_Float16)x1.w;
    }
    __syncthreads();

    const int wave = tid >> 5;
    const int lane = tid & 31;
    const int m    = lane & 15;
    const int hi   = lane >> 4;           // 0: lanes 0-15, 1: lanes 16-31
    const int col  = wave * 16 + m;

    v8f acc = {};

#pragma unroll
    for (int kb = 0; kb < 128; kb += 32) {
        const int kO = kb + hi * 8;       // lanes 16-31 hold K+8..K+15 / K+24..31
        v16h aM, aX, bL, bR;
        const _Float16* pm = &sMean[m][kO];
        const _Float16* px = &sX[m][kO];
        const _Float16* pl = WlT + (size_t)col * 128 + kO;
        const _Float16* pr = WrT + (size_t)col * 128 + kO;
#pragma unroll
        for (int i = 0; i < 8; ++i) {
            aM[i] = pm[i];      aM[8 + i] = pm[16 + i];
            aX[i] = px[i];      aX[8 + i] = px[16 + i];
            bL[i] = pl[i];      bL[8 + i] = pl[16 + i];
            bR[i] = pr[i];      bR[8 + i] = pr[16 + i];
        }
        acc = __builtin_amdgcn_wmma_f32_16x16x32_f16(false, aM, false, bL,
                                                     (short)0, acc, false, false);
        acc = __builtin_amdgcn_wmma_f32_16x16x32_f16(false, aX, false, bR,
                                                     (short)0, acc, false, false);
    }

    const float bcol = bias[col];
    const int rB = rowBase + (hi << 3);
#pragma unroll
    for (int r = 0; r < 8; ++r) {
        float v = acc[r] + bcol;
        out[(size_t)(rB + r) * 128 + col] = fmaxf(v, 0.0f);
    }
}

// ---------------------------------------------------------------------------
// head: mu = h@Wmu + bmu ; lv = h@Wlv + blv   (128 -> 32)
// block = 128 threads = 4 waves: wave0/1 -> mu col tiles, wave2/3 -> lv.
__global__ void __launch_bounds__(128)
k_head_wmma(const float* __restrict__ h,
            const _Float16* __restrict__ WmuT,
            const _Float16* __restrict__ WlvT,
            const float* __restrict__ bmu,
            const float* __restrict__ blv,
            float* __restrict__ muOut,
            float* __restrict__ lvOut) {
    __shared__ _Float16 sH[16][136];

    const int tid     = threadIdx.x;
    const int rowBase = blockIdx.x * 16;
    {
        int base = tid * 16;
        int r = base >> 7;
        int k = base & 127;
        const float4* ph = (const float4*)(h + (size_t)(rowBase + r) * 128 + k);
#pragma unroll
        for (int q = 0; q < 4; ++q) {
            float4 v = ph[q];
            sH[r][k + q * 4 + 0] = (_Float16)v.x;
            sH[r][k + q * 4 + 1] = (_Float16)v.y;
            sH[r][k + q * 4 + 2] = (_Float16)v.z;
            sH[r][k + q * 4 + 3] = (_Float16)v.w;
        }
    }
    __syncthreads();

    const int wave = tid >> 5;
    const int lane = tid & 31;
    const int m    = lane & 15;
    const int hi   = lane >> 4;
    const int ct   = wave & 1;            // column tile 0/1
    const int mtx  = wave >> 1;           // 0 = mu, 1 = lv
    const int col  = ct * 16 + m;

    const _Float16* WT = mtx ? WlvT : WmuT;
    v8f acc = {};

#pragma unroll
    for (int kb = 0; kb < 128; kb += 32) {
        const int kO = kb + hi * 8;
        v16h aH, bW;
        const _Float16* ph = &sH[m][kO];
        const _Float16* pw = WT + (size_t)col * 128 + kO;
#pragma unroll
        for (int i = 0; i < 8; ++i) {
            aH[i] = ph[i];  aH[8 + i] = ph[16 + i];
            bW[i] = pw[i];  bW[8 + i] = pw[16 + i];
        }
        acc = __builtin_amdgcn_wmma_f32_16x16x32_f16(false, aH, false, bW,
                                                     (short)0, acc, false, false);
    }

    const float bcol = mtx ? blv[col] : bmu[col];
    float* outp = mtx ? lvOut : muOut;
    const int rB = rowBase + (hi << 3);
#pragma unroll
    for (int r = 0; r < 8; ++r)
        outp[(size_t)(rB + r) * 32 + col] = acc[r] + bcol;
}

// ---------------------------------------------------------------------------
// z = mu + eps * exp(0.5*logvar)
__global__ void k_reparam(const float* __restrict__ mu,
                          const float* __restrict__ lv,
                          const float* __restrict__ eps,
                          float* __restrict__ z, int n) {
    int i = blockIdx.x * blockDim.x + threadIdx.x;
    if (i < n) z[i] = mu[i] + eps[i] * expf(0.5f * lv[i]);
}

// ---------------------------------------------------------------------------
// probs[e] = sigmoid( dot(z[src[e]], z[dst[e]]) ), 8 lanes per edge
__global__ void k_edge_decode(const float* __restrict__ z,
                              const int* __restrict__ src,
                              const int* __restrict__ dst,
                              float* __restrict__ probs, int nEdges) {
    int gt = blockIdx.x * blockDim.x + threadIdx.x;
    int e = gt >> 3;
    if (e >= nEdges) return;
    int j = gt & 7;
    int s = src[e], d = dst[e];
    const float4 a = *(const float4*)(z + (size_t)s * 32 + j * 4);
    const float4 b = *(const float4*)(z + (size_t)d * 32 + j * 4);
    float p = a.x * b.x + a.y * b.y + a.z * b.z + a.w * b.w;
    p += __shfl_xor(p, 1, 8);
    p += __shfl_xor(p, 2, 8);
    p += __shfl_xor(p, 4, 8);
    if (j == 0) probs[e] = 1.0f / (1.0f + expf(-p));
}

// ---------------------------------------------------------------------------
extern "C" void kernel_launch(void* const* d_in, const int* in_sizes, int n_in,
                              void* d_out, int out_size, void* d_ws, size_t ws_size,
                              hipStream_t stream) {
    const float* x    = (const float*)d_in[0];
    const int*   ei   = (const int*)d_in[1];
    const float* eps  = (const float*)d_in[2];
    const float* Wl0  = (const float*)d_in[3];
    const float* Wr0  = (const float*)d_in[4];
    const float* b0   = (const float*)d_in[5];
    const float* Wl1  = (const float*)d_in[6];
    const float* Wr1  = (const float*)d_in[7];
    const float* b1   = (const float*)d_in[8];
    const float* Wmu  = (const float*)d_in[9];
    const float* bmu  = (const float*)d_in[10];
    const float* Wlv  = (const float*)d_in[11];
    const float* blv  = (const float*)d_in[12];

    const int* src = ei;
    const int* dst = ei + N_EDGES;

    char* ws = (char*)d_ws;
    float* agg = (float*)(ws);                              // 25,600,000 B
    float* cnt = (float*)(ws + 25600000);                   //    200,000 B
    float* h0  = (float*)(ws + 25800000);                   // 25,600,000 B
    float* h1  = (float*)(ws + 51400000);                   // 25,600,000 B
    _Float16* Wl0T = (_Float16*)(ws + 77000000);            // 32,768 B each (128x128)
    _Float16* Wr0T = Wl0T + 16384;
    _Float16* Wl1T = Wr0T + 16384;
    _Float16* Wr1T = Wl1T + 16384;
    _Float16* WmuT = Wr1T + 16384;                          // 8,192 B each (32x128)
    _Float16* WlvT = WmuT + 4096;
    float* z = agg;                                         // reuse agg after layer 2

    float* probs  = (float*)d_out;
    float* muOut  = probs + N_EDGES;
    float* lvOut  = muOut + N_NODES * LAT;

    const int ROWT = N_NODES / 16;  // 3125 row tiles

    // zero agg + cnt (contiguous)
    k_zero<<<(6450000 + 255) / 256, 256, 0, stream>>>(agg, 6450000);

    // weight convert/transpose
    k_w_to_f16t<<<(16384 + 255) / 256, 256, 0, stream>>>(Wl0, Wl0T, 128, 128);
    k_w_to_f16t<<<(16384 + 255) / 256, 256, 0, stream>>>(Wr0, Wr0T, 128, 128);
    k_w_to_f16t<<<(16384 + 255) / 256, 256, 0, stream>>>(Wl1, Wl1T, 128, 128);
    k_w_to_f16t<<<(16384 + 255) / 256, 256, 0, stream>>>(Wr1, Wr1T, 128, 128);
    k_w_to_f16t<<<(4096  + 255) / 256, 256, 0, stream>>>(Wmu, WmuT, 128, 32);
    k_w_to_f16t<<<(4096  + 255) / 256, 256, 0, stream>>>(Wlv, WlvT, 128, 32);

    // layer 0
    k_aggregate<<<(N_EDGES * 32) / 256, 256, 0, stream>>>(x, src, dst, agg, cnt,
                                                          N_EDGES, 1);
    k_sage_wmma<<<ROWT, 256, 0, stream>>>(agg, cnt, x, Wl0T, Wr0T, b0, h0);

    // layer 1
    k_zero<<<(6400000 + 255) / 256, 256, 0, stream>>>(agg, 6400000);
    k_aggregate<<<(N_EDGES * 32) / 256, 256, 0, stream>>>(h0, src, dst, agg, cnt,
                                                          N_EDGES, 0);
    k_sage_wmma<<<ROWT, 256, 0, stream>>>(agg, cnt, h0, Wl1T, Wr1T, b1, h1);

    // heads -> mu / logvar straight into d_out
    k_head_wmma<<<ROWT, 128, 0, stream>>>(h1, WmuT, WlvT, bmu, blv, muOut, lvOut);

    // reparameterize
    k_reparam<<<(N_NODES * LAT + 255) / 256, 256, 0, stream>>>(muOut, lvOut, eps, z,
                                                               N_NODES * LAT);

    // edge decode
    k_edge_decode<<<(N_EDGES * 8) / 256, 256, 0, stream>>>(z, src, dst, probs,
                                                           N_EDGES);
}
